// SelfAttention_57827439673579
// MI455X (gfx1250) — matrix operand order repaired and checked by reading, not compiled, not run on hardware
//
#include <hip/hip_runtime.h>

typedef __attribute__((ext_vector_type(16))) _Float16 v16h;
typedef __attribute__((ext_vector_type(8)))  _Float16 v8h;
typedef __attribute__((ext_vector_type(8)))  float    v8f;

#define BATCH   16
#define NSEQ    2048
#define DHEAD   64
#define QTILE   128          // 8 waves x 16 queries
#define KTILE   64
#define LDSROW  72           // padded row stride in halfs (144 B: 16B-aligned, conflict-free)
#define TILEH   (KTILE * LDSROW)
#define NT      (NSEQ / KTILE)
// fold 1/sqrt(64) and log2(e) into Q so softmax uses raw v_exp_f32 (base-2)
#define QSCALE  0.18033688011112042f

static __device__ __forceinline__ v16h cat8(v8h lo, v8h hi) {
  return __builtin_shufflevector(lo, hi, 0,1,2,3,4,5,6,7,8,9,10,11,12,13,14,15);
}

// swap bits 3<->4 of the 6-bit in-tile key index (involution). This key order
// makes the S^T C-layout registers directly usable as P^T B-fragments.
static __device__ __forceinline__ int swap34(int i) {
  return (i & ~0x18) | (((i >> 3) & 1) << 4) | (((i >> 4) & 1) << 3);
}

#define WMMA_F16(A, Bm, C) \
  __builtin_amdgcn_wmma_f32_16x16x32_f16(false, (A), false, (Bm), (short)0, (C), false, false)

__global__ __launch_bounds__(256, 1)
void flash_attn_wmma_t(const float* __restrict__ Q,
                       const float* __restrict__ K,
                       const float* __restrict__ V,
                       float* __restrict__ O)
{
  __shared__ _Float16 sK [2 * TILEH];   // double-buffered K tile  [key][dim]
  __shared__ _Float16 sVt[2 * TILEH];   // double-buffered V^T tile [dim][dot] (swap34 key order)

  const int tid  = threadIdx.x;
  const int wv   = tid >> 5;
  const int lane = tid & 31;
  const int hl   = lane >> 4;
  const int ln   = lane & 15;
  const int b    = blockIdx.y;
  const int qt   = blockIdx.x;

  const int c0 = hl * 8;     // A-fragment chunk offset
  const int k0 = hl * 16;    // B-fragment K offset

  const float* Kb = K + (size_t)b * NSEQ * DHEAD;
  const float* Vb = V + (size_t)b * NSEQ * DHEAD;

  // staging maps: K -> 4 thr/row, 16 dims each; V -> 2 keys x 8 dims (paired b32 stores)
  const int krow  = tid >> 2,        kcol = (tid & 3) * 16;
  const int vkey  = (tid & 31) * 2,  vdim = (tid >> 5) * 8;
  const int vkeyp = swap34(vkey);    // even; swap34 preserves bit0 so (vkey,vkey+1) stay adjacent

  auto stage = [&](int kt, int bo, bool pf) {
    const float* kr = Kb + (size_t)(kt * KTILE + krow) * DHEAD + kcol;
    #pragma unroll
    for (int i = 0; i < 16; i += 8) {
      v8h hk;
      #pragma unroll
      for (int j = 0; j < 8; j += 2) {
        auto t = __builtin_amdgcn_cvt_pkrtz(kr[i + j], kr[i + j + 1]);
        hk[j] = t[0]; hk[j + 1] = t[1];
      }
      *(v8h*)&sK[bo + krow * LDSROW + kcol + i] = hk;
    }
    const float* v0 = Vb + (size_t)(kt * KTILE + vkey) * DHEAD + vdim;
    const float* v1 = v0 + DHEAD;
    #pragma unroll
    for (int j = 0; j < 8; ++j) {
      auto t = __builtin_amdgcn_cvt_pkrtz(v0[j], v1[j]);   // (key, key+1) pair for dim vdim+j
      *(decltype(t)*)&sVt[bo + (vdim + j) * LDSROW + vkeyp] = t;
    }
    if (pf) {
      __builtin_prefetch(kr + KTILE * DHEAD, 0, 0);
      __builtin_prefetch(v0 + KTILE * DHEAD, 0, 0);
    }
  };

  // ---- Q^T resident B-fragments (S^T = K · Q^T); lane ln owns query column qrow
  const int    qrow = qt * QTILE + wv * 16 + ln;
  const float* Qp   = Q + ((size_t)b * NSEQ + qrow) * DHEAD;
  v16h qb[2];
  #pragma unroll
  for (int kk = 0; kk < 2; ++kk)
    #pragma unroll
    for (int j = 0; j < 16; ++j)
      qb[kk][j] = (_Float16)(Qp[kk * 32 + k0 + j] * QSCALE);

  v8f   o[4] = {};
  float m = -1e30f, l = 0.0f;

  stage(0, 0, true);

  for (int kt = 0; kt < NT; ++kt) {
    const int bo = (kt & 1) * TILEH;
    __syncthreads();                                   // buf[kt&1] staged, other buf free
    if (kt + 1 < NT) stage(kt + 1, TILEH - bo, kt + 2 < NT);

    // ---- S^T = K · (Q·qscale)^T : 4 key-row tiles, K-dim 64 (2x K32 WMMA)
    v8f s[4];
    #pragma unroll
    for (int t = 0; t < 4; ++t) {
      const _Float16* kp = &sK[bo + (t * 16 + ln) * LDSROW];
      v16h a0 = cat8(*(const v8h*)(kp + k0),      *(const v8h*)(kp + k0 + 8));
      v16h a1 = cat8(*(const v8h*)(kp + 32 + k0), *(const v8h*)(kp + 32 + k0 + 8));
      v8f acc = {};
      acc  = WMMA_F16(a0, qb[0], acc);
      s[t] = WMMA_F16(a1, qb[1], acc);
    }

    // ---- base-2 online softmax; per-lane stats + one xor-16 merge
    v8f vm;
    #pragma unroll
    for (int r = 0; r < 8; ++r)
      vm[r] = fmaxf(fmaxf(s[0][r], s[1][r]), fmaxf(s[2][r], s[3][r]));
    float mx = fmaxf(fmaxf(fmaxf(vm[0], vm[1]), fmaxf(vm[2], vm[3])),
                     fmaxf(fmaxf(vm[4], vm[5]), fmaxf(vm[6], vm[7])));
    mx = fmaxf(mx, __shfl_xor(mx, 16, 32));

    const float mnew  = fmaxf(m, mx);
    const float alpha = __builtin_amdgcn_exp2f(m - mnew);
    m = mnew;

    v8f psv = {};
    #pragma unroll
    for (int t = 0; t < 4; ++t)
      #pragma unroll
      for (int r = 0; r < 8; ++r) {
        float p = __builtin_amdgcn_exp2f(s[t][r] - mnew);
        s[t][r] = p;
        psv[r] += p;
      }
    float ps = ((psv[0] + psv[1]) + (psv[2] + psv[3])) +
               ((psv[4] + psv[5]) + (psv[6] + psv[7]));
    ps += __shfl_xor(ps, 16, 32);
    l = l * alpha + ps;

    #pragma unroll
    for (int t = 0; t < 4; ++t)
      #pragma unroll
      for (int r = 0; r < 8; ++r) o[t][r] *= alpha;

    // ---- P^T B-fragments: in-lane register identity + packed f16 converts
    v16h pb[2];
    #pragma unroll
    for (int kk = 0; kk < 2; ++kk)
      #pragma unroll
      for (int j = 0; j < 8; j += 2) {
        auto lo = __builtin_amdgcn_cvt_pkrtz(s[2 * kk][j],     s[2 * kk][j + 1]);
        auto hi = __builtin_amdgcn_cvt_pkrtz(s[2 * kk + 1][j], s[2 * kk + 1][j + 1]);
        pb[kk][j]     = lo[0]; pb[kk][j + 1]     = lo[1];
        pb[kk][8 + j] = hi[0]; pb[kk][8 + j + 1] = hi[1];
      }

    // ---- O^T += V^T · P^T
    #pragma unroll
    for (int dt = 0; dt < 4; ++dt) {
      const _Float16* vp = &sVt[bo + (dt * 16 + ln) * LDSROW];
      v16h a0 = cat8(*(const v8h*)(vp + k0),      *(const v8h*)(vp + k0 + 8));
      v16h a1 = cat8(*(const v8h*)(vp + 32 + k0), *(const v8h*)(vp + 32 + k0 + 8));
      o[dt] = WMMA_F16(a0, pb[0], o[dt]);
      o[dt] = WMMA_F16(a1, pb[1], o[dt]);
    }
  }

  // ---- normalize + contiguous vector stores (each lane owns one query row)
  const float inv = 1.0f / l;
  float* Op = O + ((size_t)b * NSEQ + qrow) * DHEAD;
  #pragma unroll
  for (int dt = 0; dt < 4; ++dt) {
    v8f r;
    #pragma unroll
    for (int j = 0; j < 8; ++j) r[j] = o[dt][j] * inv;
    *(v8f*)(Op + dt * 16 + hl * 8) = r;
  }
}

extern "C" void kernel_launch(void* const* d_in, const int* in_sizes, int n_in,
                              void* d_out, int out_size, void* d_ws, size_t ws_size,
                              hipStream_t stream) {
  (void)in_sizes; (void)n_in; (void)out_size; (void)d_ws; (void)ws_size;
  const float* q = (const float*)d_in[0];
  const float* k = (const float*)d_in[1];
  const float* v = (const float*)d_in[2];
  float* out = (float*)d_out;
  dim3 grid(NSEQ / QTILE, BATCH);
  dim3 block(256);
  hipLaunchKernelGGL(flash_attn_wmma_t, grid, block, 0, stream, q, k, v, out);
}
